// Block_23622320128656
// MI455X (gfx1250) — matrix operand order repaired and checked by reading, not compiled, not run on hardware
//
#include <hip/hip_runtime.h>
#include <stdint.h>

#define C_EMBD  1024
#define T_SEQ   2048
#define N_BATCH 4
#define NHEADS  16
#define HEAD_D  64
#define N_ROWS  (N_BATCH * T_SEQ)   // 8192
#define FFN     4096

typedef __bf16 bf16_t;
typedef bf16_t        v16bf __attribute__((ext_vector_type(16)));
typedef float         v8f   __attribute__((ext_vector_type(8)));
typedef unsigned int  u32x4 __attribute__((ext_vector_type(4)));
typedef float         f32x4 __attribute__((ext_vector_type(4)));

union Frag {
    u32x4          q[2];
    unsigned short s[16];
    v16bf          v;
};

// Async memory->LDS path (gfx1250 GLOBAL_LOAD_ASYNC_TO_LDS_B128, ASYNCcnt).
// Builtin signature (confirmed via hipcc): (v4i AS1*, v4i AS3*, Imm, Imm).
#if defined(__AMDGCN__) && __has_builtin(__builtin_amdgcn_global_load_async_to_lds_b128) && \
    __has_builtin(__builtin_amdgcn_s_wait_asynccnt)
#define HAVE_ASYNC_LDS 1
typedef int v4i_t __attribute__((vector_size(16)));
typedef __attribute__((address_space(1))) v4i_t* gas_v4i;
typedef __attribute__((address_space(3))) v4i_t* las_v4i;
#define AS1(p) ((gas_v4i)(uintptr_t)(p))
#define AS3(p) ((las_v4i)(uintptr_t)(p))
#else
#define HAVE_ASYNC_LDS 0
#endif

__device__ __forceinline__ unsigned short f32_to_bf16(float f) {
    union { float f; uint32_t u; } cv;
    cv.f = f;
    uint32_t u   = cv.u;
    uint32_t lsb = (u >> 16) & 1u;
    u += 0x7fffu + lsb;               // round to nearest even
    return (unsigned short)(u >> 16);
}

// ---------------------------------------------------------------------------
// Transpose + fp32->bf16 convert for weights:  Wt[n][k] = bf16(W[k][n]).
// ---------------------------------------------------------------------------
__global__ __launch_bounds__(256) void k_transpose_bf16(
    const float* __restrict__ W, unsigned short* __restrict__ Wt, int K, int N) {
    __shared__ unsigned short tile[16][17];
    const int tx = threadIdx.x & 15;
    const int ty = threadIdx.x >> 4;
    const int n0 = blockIdx.x * 16;
    const int k0 = blockIdx.y * 16;
    tile[ty][tx] = f32_to_bf16(W[(size_t)(k0 + ty) * N + (n0 + tx)]);
    __syncthreads();
    Wt[(size_t)(n0 + ty) * K + (k0 + tx)] = tile[tx][ty];
}

// ---------------------------------------------------------------------------
// LayerNorm (fp32 in) -> bf16 out.  One 256-thread block per row of 1024.
// ---------------------------------------------------------------------------
__global__ __launch_bounds__(256) void k_layernorm_bf16(
    const float* __restrict__ x, const float* __restrict__ g,
    const float* __restrict__ beta, unsigned short* __restrict__ out) {
    __shared__ float red[256];
    const int row = blockIdx.x;
    const int tid = threadIdx.x;
    const float* xr = x + (size_t)row * C_EMBD;
    f32x4 v = ((const f32x4*)xr)[tid];
    red[tid] = v[0] + v[1] + v[2] + v[3];
    __syncthreads();
#pragma unroll
    for (int off = 128; off > 0; off >>= 1) {
        if (tid < off) red[tid] += red[tid + off];
        __syncthreads();
    }
    const float mu = red[0] * (1.0f / C_EMBD);
    __syncthreads();
    f32x4 d = v - mu;
    red[tid] = d[0] * d[0] + d[1] * d[1] + d[2] * d[2] + d[3] * d[3];
    __syncthreads();
#pragma unroll
    for (int off = 128; off > 0; off >>= 1) {
        if (tid < off) red[tid] += red[tid + off];
        __syncthreads();
    }
    const float rstd = rsqrtf(red[0] * (1.0f / C_EMBD) + 1e-5f);
    const int c = tid * 4;
#pragma unroll
    for (int j = 0; j < 4; ++j)
        out[(size_t)row * C_EMBD + c + j] =
            f32_to_bf16(d[j] * rstd * g[c + j] + beta[c + j]);
}

// ---------------------------------------------------------------------------
// bf16 WMMA GEMM: C(MxN) = A(MxK) * Bt(NxK)^T, compile-time epilogue.
// 128x128 block tile, BK=32, 8 waves (4x2), wave tile 32x64.
// Async double-buffered LDS pipeline: issue K-step i+1 before computing i.
// OUT_BF16_VT: store bf16 transposed per-batch: out[b][col][t]  (for V).
// ---------------------------------------------------------------------------
#define BM 128
#define BN 128
#define BK 32

template <bool HAS_BIAS, bool DO_RELU, bool HAS_RES, bool OUT_F32,
          bool OUT_BF16, bool OUT_BF16_VT>
__global__ __launch_bounds__(256) void k_gemm_bf16(
    const unsigned short* __restrict__ A,    // M x K
    const unsigned short* __restrict__ Bt,   // N x K (pre-transposed weights)
    int M, int N, int K,
    const float* __restrict__ bias, const float* __restrict__ residual,
    float* __restrict__ outf, unsigned short* __restrict__ outb) {
#if HAVE_ASYNC_LDS
    __shared__ __align__(16) unsigned short As[2][BM][BK];  // 16 KB
    __shared__ __align__(16) unsigned short Bs[2][BM][BK];  // 16 KB
#else
    __shared__ __align__(16) unsigned short As[1][BM][BK];
    __shared__ __align__(16) unsigned short Bs[1][BM][BK];
#endif

    const int tid  = threadIdx.x;
    const int wave = tid >> 5;
    const int lane = tid & 31;
    const int nl   = lane & 15;
    const int hi   = lane >> 4;
    const int wm   = wave >> 1;   // 0..3
    const int wn   = wave & 1;    // 0..1
    const int bm   = blockIdx.y * BM;
    const int bn   = blockIdx.x * BN;

    v8f acc[2][4] = {};

    const int ar = tid >> 1;          // 0..127 (row of either tile)
    const int ac = (tid & 1) << 4;    // 0,16   (k offset)
    const unsigned short* gA = A + (size_t)(bm + ar) * K + ac;
    const unsigned short* gB = Bt + (size_t)(bn + ar) * K + ac;

    auto compute_tile = [&](int buf) {
        Frag afrag[2];
#pragma unroll
        for (int mi = 0; mi < 2; ++mi) {
            const int row = wm * 32 + mi * 16 + nl;
            afrag[mi].q[0] = *(const u32x4*)&As[buf][row][hi * 8];
            afrag[mi].q[1] = *(const u32x4*)&As[buf][row][16 + hi * 8];
        }
#pragma unroll
        for (int ni = 0; ni < 4; ++ni) {
            Frag bfrag;
            const int col = wn * 64 + ni * 16 + nl;
            bfrag.q[0] = *(const u32x4*)&Bs[buf][col][hi * 8];
            bfrag.q[1] = *(const u32x4*)&Bs[buf][col][16 + hi * 8];
#pragma unroll
            for (int mi = 0; mi < 2; ++mi) {
                acc[mi][ni] = __builtin_amdgcn_wmma_f32_16x16x32_bf16(
                    false, afrag[mi].v, false, bfrag.v, (short)0, acc[mi][ni],
                    false, false);
            }
        }
    };

#if HAVE_ASYNC_LDS
    auto issue = [&](int k0, int buf) {
        __builtin_amdgcn_global_load_async_to_lds_b128(AS1(gA + k0),     AS3(&As[buf][ar][ac]),     0, 0);
        __builtin_amdgcn_global_load_async_to_lds_b128(AS1(gA + k0 + 8), AS3(&As[buf][ar][ac + 8]), 0, 0);
        __builtin_amdgcn_global_load_async_to_lds_b128(AS1(gB + k0),     AS3(&Bs[buf][ar][ac]),     0, 0);
        __builtin_amdgcn_global_load_async_to_lds_b128(AS1(gB + k0 + 8), AS3(&Bs[buf][ar][ac + 8]), 0, 0);
    };
    issue(0, 0);
    int buf = 0;
    for (int k0 = 0; k0 < K; k0 += BK) {
        if (k0 + BK < K) {
            issue(k0 + BK, buf ^ 1);          // overlap next copy with compute
            __builtin_amdgcn_s_wait_asynccnt(4);  // oldest 4 (this tile) done
        } else {
            __builtin_amdgcn_s_wait_asynccnt(0);
        }
        __syncthreads();     // all waves' async writes for 'buf' visible
        compute_tile(buf);
        __syncthreads();     // all reads of 'buf' done before it is refilled
        buf ^= 1;
    }
#else
    for (int k0 = 0; k0 < K; k0 += BK) {
        u32x4 a0 = ((const u32x4*)(gA + k0))[0];
        u32x4 a1 = ((const u32x4*)(gA + k0))[1];
        u32x4 b0 = ((const u32x4*)(gB + k0))[0];
        u32x4 b1 = ((const u32x4*)(gB + k0))[1];
        __syncthreads();
        *(u32x4*)&As[0][ar][ac]     = a0;
        *(u32x4*)&As[0][ar][ac + 8] = a1;
        *(u32x4*)&Bs[0][ar][ac]     = b0;
        *(u32x4*)&Bs[0][ar][ac + 8] = b1;
        __syncthreads();
        compute_tile(0);
    }
#endif

#pragma unroll
    for (int mi = 0; mi < 2; ++mi) {
#pragma unroll
        for (int ni = 0; ni < 4; ++ni) {
            const int col = bn + wn * 64 + ni * 16 + nl;
            const float bv = HAS_BIAS ? bias[col] : 0.0f;
#pragma unroll
            for (int r = 0; r < 8; ++r) {
                const int row = bm + wm * 32 + mi * 16 + r + hi * 8;
                float v = acc[mi][ni][r] + bv;
                if (DO_RELU) v = fmaxf(v, 0.0f);
                if (HAS_RES) v += residual[(size_t)row * N + col];
                if (OUT_F32) outf[(size_t)row * N + col] = v;
                if (OUT_BF16) outb[(size_t)row * N + col] = f32_to_bf16(v);
                if (OUT_BF16_VT) {
                    const int bb = row >> 11;           // row / T_SEQ
                    const int tt = row & (T_SEQ - 1);   // row % T_SEQ
                    outb[((size_t)bb * C_EMBD + col) * T_SEQ + tt] =
                        f32_to_bf16(v);
                }
            }
        }
    }
}

// ---------------------------------------------------------------------------
// One 32-key flash-attention step for a wave owning 16 query rows.
// MASKED=true only for the single diagonal step per wave.
// ---------------------------------------------------------------------------
template <bool MASKED>
__device__ __forceinline__ void attn_step(
    int k0, int qr0, int nl, int hi, const Frag* qa,
    const unsigned short* __restrict__ kbase,   // Kt + batch/head base
    const unsigned short* __restrict__ vbase,   // Vt + batch/head base (d-major)
    unsigned short (*Pbw)[32],                  // per-wave [16][32] LDS tile
    float* mrun, float* lrun, v8f* oacc) {
    const float scale = 0.03125f;  // C^-0.5 = 1/32

    // scores S = Q * K^T : two 16x16 tiles (keys k0..+15, k0+16..+31)
    v8f s0 = {}, s1 = {};
    {
        const unsigned short* kp0 = kbase + (size_t)(k0 + nl) * C_EMBD;
        const unsigned short* kp1 = kbase + (size_t)(k0 + 16 + nl) * C_EMBD;
#pragma unroll
        for (int s = 0; s < 2; ++s) {
            Frag kb0, kb1;
            kb0.q[0] = *(const u32x4*)(kp0 + s * 32 + hi * 8);
            kb0.q[1] = *(const u32x4*)(kp0 + s * 32 + 16 + hi * 8);
            kb1.q[0] = *(const u32x4*)(kp1 + s * 32 + hi * 8);
            kb1.q[1] = *(const u32x4*)(kp1 + s * 32 + 16 + hi * 8);
            s0 = __builtin_amdgcn_wmma_f32_16x16x32_bf16(
                false, qa[s].v, false, kb0.v, (short)0, s0, false, false);
            s1 = __builtin_amdgcn_wmma_f32_16x16x32_bf16(
                false, qa[s].v, false, kb1.v, (short)0, s1, false, false);
        }
    }

    // scale (+ causal mask on the diagonal step only); per-row step max
    float mstep[8];
#pragma unroll
    for (int r = 0; r < 8; ++r) {
        if (MASKED) {
            const int qi = qr0 + r + hi * 8;
            s0[r] = ((k0 + nl) <= qi) ? s0[r] * scale : -3.0e38f;
            s1[r] = ((k0 + 16 + nl) <= qi) ? s1[r] * scale : -3.0e38f;
        } else {
            s0[r] *= scale;
            s1[r] *= scale;
        }
        mstep[r] = fmaxf(s0[r], s1[r]);
    }
#pragma unroll
    for (int off = 1; off < 16; off <<= 1)
#pragma unroll
        for (int r = 0; r < 8; ++r)
            mstep[r] = fmaxf(mstep[r], __shfl_xor(mstep[r], off, 32));

    float fac[8], rsum[8];
#pragma unroll
    for (int r = 0; r < 8; ++r) {
        const float mn = fmaxf(mrun[r], mstep[r]);
        fac[r]  = __expf(mrun[r] - mn);
        mrun[r] = mn;
        s0[r]   = __expf(s0[r] - mn);
        s1[r]   = __expf(s1[r] - mn);
        rsum[r] = s0[r] + s1[r];
    }
#pragma unroll
    for (int off = 1; off < 16; off <<= 1)
#pragma unroll
        for (int r = 0; r < 8; ++r)
            rsum[r] += __shfl_xor(rsum[r], off, 32);
#pragma unroll
    for (int r = 0; r < 8; ++r) lrun[r] = lrun[r] * fac[r] + rsum[r];
#pragma unroll
    for (int ni = 0; ni < 4; ++ni)
#pragma unroll
        for (int r = 0; r < 8; ++r) oacc[ni][r] *= fac[r];

    // P: C-layout f32 -> bf16 row-major in LDS -> reload in A layout
#pragma unroll
    for (int r = 0; r < 8; ++r) {
        Pbw[r + hi * 8][nl]      = f32_to_bf16(s0[r]);
        Pbw[r + hi * 8][16 + nl] = f32_to_bf16(s1[r]);
    }
    Frag pf;
    pf.q[0] = *(const u32x4*)&Pbw[nl][hi * 16];
    pf.q[1] = *(const u32x4*)&Pbw[nl][hi * 16 + 8];

    // O += P * V : V fragments load contiguously from transposed Vt
#pragma unroll
    for (int ni = 0; ni < 4; ++ni) {
        Frag vf;
        const unsigned short* vp =
            vbase + (size_t)(ni * 16 + nl) * T_SEQ + k0 + hi * 16;
        vf.q[0] = ((const u32x4*)vp)[0];
        vf.q[1] = ((const u32x4*)vp)[1];
        oacc[ni] = __builtin_amdgcn_wmma_f32_16x16x32_bf16(
            false, pf.v, false, vf.v, (short)0, oacc[ni], false, false);
    }
}

// ---------------------------------------------------------------------------
// Causal flash attention, bf16 WMMA, f32 online softmax.
// Q,K in [b, t, h*64+d] bf16; V pre-transposed: Vt[b][h*64+d][t] bf16.
// grid = (B*H, T/64), 128 threads (4 waves); each wave owns 16 query rows.
// ---------------------------------------------------------------------------
__global__ __launch_bounds__(128) void k_attention(
    const unsigned short* __restrict__ Q, const unsigned short* __restrict__ Kt,
    const unsigned short* __restrict__ Vt, unsigned short* __restrict__ O) {
    __shared__ __align__(16) unsigned short Pb[4][16][32];  // per-wave P tile

    const int tid  = threadIdx.x;
    const int w    = tid >> 5;
    const int lane = tid & 31;
    const int nl   = lane & 15;
    const int hi   = lane >> 4;

    const int bh = blockIdx.x;        // 0..63
    const int b  = bh >> 4;
    const int h  = bh & 15;
    const int qr0 = blockIdx.y * 64 + w * 16;

    const unsigned short* kbase =
        Kt + (size_t)b * T_SEQ * C_EMBD + (size_t)h * HEAD_D;
    const unsigned short* vbase =
        Vt + ((size_t)b * C_EMBD + (size_t)h * HEAD_D) * T_SEQ;

    // Q fragments (A operand), two K-slabs of 32 over head dim 64
    Frag qa[2];
    {
        const unsigned short* qp = Q + (size_t)b * T_SEQ * C_EMBD +
                                   (size_t)h * HEAD_D +
                                   (size_t)(qr0 + nl) * C_EMBD;
#pragma unroll
        for (int s = 0; s < 2; ++s) {
            qa[s].q[0] = *(const u32x4*)(qp + s * 32 + hi * 8);
            qa[s].q[1] = *(const u32x4*)(qp + s * 32 + 16 + hi * 8);
        }
    }

    float mrun[8], lrun[8];
#pragma unroll
    for (int r = 0; r < 8; ++r) { mrun[r] = -3.0e38f; lrun[r] = 0.0f; }
    v8f oacc[4] = {};

    // bulk steps: fully below the diagonal, no masking needed
    int k0 = 0;
    for (; k0 + 32 <= qr0; k0 += 32)
        attn_step<false>(k0, qr0, nl, hi, qa, kbase, vbase, Pb[w],
                         mrun, lrun, oacc);
    // single diagonal step (k0 <= qr0+15 < k0+32): masked
    const int qlast = qr0 + 15;
    for (; k0 <= qlast; k0 += 32)
        attn_step<true>(k0, qr0, nl, hi, qa, kbase, vbase, Pb[w],
                        mrun, lrun, oacc);

    // normalize and store bf16 attention output
#pragma unroll
    for (int r = 0; r < 8; ++r) {
        const float inv = 1.0f / lrun[r];
        unsigned short* op = O + (size_t)b * T_SEQ * C_EMBD +
                             (size_t)h * HEAD_D +
                             (size_t)(qr0 + r + hi * 8) * C_EMBD;
#pragma unroll
        for (int ni = 0; ni < 4; ++ni)
            op[ni * 16 + nl] = f32_to_bf16(oacc[ni][r] * inv);
    }
}

// ---------------------------------------------------------------------------
extern "C" void kernel_launch(void* const* d_in, const int* in_sizes, int n_in,
                              void* d_out, int out_size, void* d_ws,
                              size_t ws_size, hipStream_t stream) {
    const float* x     = (const float*)d_in[0];
    const float* Wq    = (const float*)d_in[1];
    const float* Wk    = (const float*)d_in[2];
    const float* Wv    = (const float*)d_in[3];
    const float* Wo    = (const float*)d_in[4];
    const float* bo    = (const float*)d_in[5];
    const float* W1    = (const float*)d_in[6];
    const float* b1    = (const float*)d_in[7];
    const float* W2    = (const float*)d_in[8];
    const float* b2    = (const float*)d_in[9];
    const float* ln1_g = (const float*)d_in[10];
    const float* ln1_b = (const float*)d_in[11];
    const float* ln2_g = (const float*)d_in[12];
    const float* ln2_b = (const float*)d_in[13];
    float* out = (float*)d_out;

    const size_t SZ_W   = (size_t)C_EMBD * C_EMBD * 2;  // 2 MB
    const size_t SZ_W1  = (size_t)C_EMBD * FFN * 2;     // 8 MB
    const size_t SZ_ACT = (size_t)N_ROWS * C_EMBD * 2;  // 16 MB

    char* ws = (char*)d_ws;
    unsigned short* wq_t   = (unsigned short*)(ws);
    unsigned short* wk_t   = (unsigned short*)(ws + SZ_W);
    unsigned short* wv_t   = (unsigned short*)(ws + 2 * SZ_W);
    unsigned short* wo_t   = (unsigned short*)(ws + 3 * SZ_W);
    unsigned short* w1_t   = (unsigned short*)(ws + 4 * SZ_W);
    unsigned short* w2_t   = (unsigned short*)(ws + 4 * SZ_W + SZ_W1);
    char* act              = ws + 4 * SZ_W + 2 * SZ_W1;
    unsigned short* h_b    = (unsigned short*)(act);                // reused: h2
    unsigned short* q_b    = (unsigned short*)(act + SZ_ACT);
    unsigned short* k_b    = (unsigned short*)(act + 2 * SZ_ACT);
    unsigned short* vt_b   = (unsigned short*)(act + 3 * SZ_ACT);
    unsigned short* attn_b = (unsigned short*)(act + 4 * SZ_ACT);
    unsigned short* a1_b   = q_b;  // overlay: q/k/vt/attn dead when a1 live

    // 1) weights -> bf16, transposed to N x K
    auto tconv = [&](const float* src, unsigned short* dst, int K, int N) {
        k_transpose_bf16<<<dim3(N / 16, K / 16), 256, 0, stream>>>(src, dst, K, N);
    };
    tconv(Wq, wq_t, C_EMBD, C_EMBD);
    tconv(Wk, wk_t, C_EMBD, C_EMBD);
    tconv(Wv, wv_t, C_EMBD, C_EMBD);
    tconv(Wo, wo_t, C_EMBD, C_EMBD);
    tconv(W1, w1_t, C_EMBD, FFN);
    tconv(W2, w2_t, FFN, C_EMBD);

    // 2) h = LN1(x) (bf16)
    k_layernorm_bf16<<<N_ROWS, 256, 0, stream>>>(x, ln1_g, ln1_b, h_b);

    // 3) q/k = h @ Wq/Wk (bf16); v = h @ Wv stored transposed (Vt[b][c][t])
    dim3 gC(C_EMBD / BN, N_ROWS / BM);  // (8, 64)
    k_gemm_bf16<false, false, false, false, true, false>
        <<<gC, 256, 0, stream>>>(h_b, wq_t, N_ROWS, C_EMBD, C_EMBD,
                                 nullptr, nullptr, nullptr, q_b);
    k_gemm_bf16<false, false, false, false, true, false>
        <<<gC, 256, 0, stream>>>(h_b, wk_t, N_ROWS, C_EMBD, C_EMBD,
                                 nullptr, nullptr, nullptr, k_b);
    k_gemm_bf16<false, false, false, false, false, true>
        <<<gC, 256, 0, stream>>>(h_b, wv_t, N_ROWS, C_EMBD, C_EMBD,
                                 nullptr, nullptr, nullptr, vt_b);

    // 4) causal attention
    k_attention<<<dim3(N_BATCH * NHEADS, T_SEQ / 64), 128, 0, stream>>>(
        q_b, k_b, vt_b, attn_b);

    // 5) x1 = x + attn @ Wo + bo   (f32, into d_out)
    k_gemm_bf16<true, false, true, true, false, false>
        <<<gC, 256, 0, stream>>>(attn_b, wo_t, N_ROWS, C_EMBD, C_EMBD,
                                 bo, x, out, nullptr);

    // 6) h2 = LN2(x1)
    k_layernorm_bf16<<<N_ROWS, 256, 0, stream>>>(out, ln2_g, ln2_b, h_b);

    // 7) a1 = relu(h2 @ W1 + b1)  (bf16)
    dim3 gF(FFN / BN, N_ROWS / BM);  // (32, 64)
    k_gemm_bf16<true, true, false, false, true, false>
        <<<gF, 256, 0, stream>>>(h_b, w1_t, N_ROWS, FFN, C_EMBD,
                                 b1, nullptr, nullptr, a1_b);

    // 8) out = x1 + a1 @ W2 + b2  (f32, in-place over x1 in d_out)
    k_gemm_bf16<true, false, true, true, false, false>
        <<<gC, 256, 0, stream>>>(a1_b, w2_t, N_ROWS, C_EMBD, FFN,
                                 b2, out, out, nullptr);

    (void)in_sizes; (void)n_in; (void)out_size; (void)ws_size;
}
